// Decoder_24799141167169
// MI455X (gfx1250) — matrix operand order repaired and checked by reading, not compiled, not run on hardware
//
#include <hip/hip_runtime.h>
#include <hip/hip_bf16.h>

// ---------------------------------------------------------------------------
// MI455X (gfx1250) fused LSTM-warp + weight-normalized SDF decoder.
//
//  * all GEMMs via v_wmma_f32_16x16x32_bf16 (bf16 in, f32 accumulate)
//  * weights pre-converted to bf16 B-fragments in d_ws (L2-resident)
//  * h / c / xyz resident in LDS across all 8 recurrent steps
//  * 8 waves / workgroup (2 per SIMD32) for latency hiding:
//      wave = (row-slice 0..3) x (column-half 0..1)
//  * LDS row strides padded so stride(DW) % 64 == 4*odd -> conflict-free-ish
// ---------------------------------------------------------------------------

typedef __attribute__((ext_vector_type(16))) __bf16       bf16x16;
typedef __attribute__((ext_vector_type(8)))  float        f32x8;
typedef __attribute__((ext_vector_type(4)))  unsigned int u32x4;

union FragU { u32x4 q[2]; bf16x16 v; };

#define HDIM   512
#define DDIM   259
#define MROWS  64           // rows per workgroup
#define NSTEPS 8

// padded LDS row strides (elements)
#define XSTR   296          // xcat bf16 (>= 288)      592 B = 37*16
#define HSTR   520          // h    bf16 (>= 512)     1040 B = 65*16
#define CSTR   516          // c    f32  (>= 512)
#define DSTR   264          // decoder bufs bf16 (>= 256)
#define PSTR   40           // p buffer bf16 (>= 32)

// ---- LDS layout (bytes) ----------------------------------------------------
#define XCAT_B   (MROWS * XSTR * 2)           // 37888
#define HBUF_B   (MROWS * HSTR * 2)           // 66560
#define CBUF_B   (MROWS * CSTR * 4)           // 132096
#define XYZ_B    (1024)                       // f32 [64][3] (padded)
#define SCR_B    (4 * 256 * 4)                // f32 per row-wave 16x16
#define SMEM_TOTAL (XCAT_B + HBUF_B + CBUF_B + XYZ_B + SCR_B)  // 241664 B

__device__ inline f32x8 splat8(float x) {
  f32x8 v;
#pragma unroll
  for (int i = 0; i < 8; ++i) v[i] = x;
  return v;
}

__device__ inline float sigm(float x) { return 1.0f / (1.0f + __expf(-x)); }

// A-fragment (16x32 bf16, ISA 7.12.2): per lane two contiguous 8-elem runs.
// lanes 0-15 : M=lane,    K = kb*32 + {0..7, 16..23}
// lanes 16-31: M=lane-16, K = kb*32 + {8..15, 24..31}
__device__ inline bf16x16 load_a(const __bf16* base, int row0, int strideK,
                                 int kb, int lane) {
  const int m  = lane & 15;
  const int k0 = kb * 32 + ((lane & 16) ? 8 : 0);
  const __bf16* p = base + (row0 + m) * strideK + k0;
  FragU u;
  u.q[0] = *(const u32x4*)(p);
  u.q[1] = *(const u32x4*)(p + 16);
  return u.v;
}

// B-fragment: 512 bf16 per 32x16 block, lane's 16 values contiguous (32 B).
__device__ inline bf16x16 load_b(const __bf16* frag, int blk, int lane) {
  return *(const bf16x16*)(frag + ((size_t)blk << 9) + (lane << 4));
}

__device__ inline f32x8 wmma_bf16(bf16x16 a, bf16x16 b, f32x8 c) {
  return __builtin_amdgcn_wmma_f32_16x16x32_bf16(false, a, false, b,
                                                 (short)0, c, false, false);
}

// ---------------------------------------------------------------------------
// Precompute kernels
// ---------------------------------------------------------------------------

// Per-row weight-norm scale: s = g[row] / ||v[row]||  (5 layers x 256 rows)
__global__ void rownorm_kernel(const float* v0, const float* g0,
                               const float* v1, const float* g1,
                               const float* v2, const float* g2,
                               const float* v3, const float* g3,
                               const float* v4, const float* g4,
                               float* s) {
  const int row   = blockIdx.x & 255;
  const int layer = blockIdx.x >> 8;
  const float* v; const float* g; int din;
  switch (layer) {
    case 0: v = v0; g = g0; din = 3;   break;
    case 1: v = v1; g = g1; din = 256; break;
    case 2: v = v2; g = g2; din = 256; break;
    case 3: v = v3; g = g3; din = 256; break;
    default: v = v4; g = g4; din = 256; break;
  }
  const int lane = threadIdx.x;
  float sum = 0.0f;
  for (int k = lane; k < din; k += 32) {
    float x = v[(size_t)row * din + k];
    sum += x * x;
  }
#pragma unroll
  for (int off = 16; off; off >>= 1) sum += __shfl_xor(sum, off, 32);
  if (lane == 0) s[layer * 256 + row] = g[row] / sqrtf(sum);
}

// W (N_out x K_in, row-major f32) -> bf16 B-fragments, block (kb,nb) at
// dst + (kb*NB + nb)*512, element order matching load_b/WMMA B layout.
__global__ void fragw_kernel(const float* __restrict__ W,
                             const float* __restrict__ scale,
                             __bf16* __restrict__ dst,
                             int N, int K, int KB, int NB) {
  const int t = blockIdx.x * blockDim.x + threadIdx.x;
  if (t >= KB * NB * 512) return;
  const int b  = t >> 9,  e  = t & 511;
  const int kb = b / NB,  nb = b % NB;
  const int l  = e >> 4,  r  = e & 15;
  const int vv = r >> 1,  hh = r & 1;
  const int n  = nb * 16 + (l & 15);
  const int k  = kb * 32 + ((l & 16) ? 8 : 0) + (r & 7) + ((r & 8) ? 16 : 0);
  float val = 0.0f;
  if (n < N && k < K) {
    val = W[(size_t)n * K + k];
    if (scale) val *= scale[n];
  }
  dst[t] = (__bf16)val;
}

__global__ void biasgate_kernel(const float* bih, const float* bhh, float* bg) {
  const int t = blockIdx.x * blockDim.x + threadIdx.x;
  if (t < 2048) bg[t] = bih[t] + bhh[t];
}

// ---------------------------------------------------------------------------
// Fused main kernel
// ---------------------------------------------------------------------------

// one decoder layer over 8 output tiles [nb0, nb0+8) for this wave's 16 rows
__device__ inline void decoder_layer(const __bf16* bin, int strideK, int KB,
                                     const __bf16* w, const float* bias,
                                     __bf16* bout, int nb0,
                                     int lane, int wr, int rbase, int colL) {
#pragma unroll 1
  for (int i = 0; i < 8; ++i) {
    const int nb = nb0 + i;
    f32x8 acc = splat8(bias[nb * 16 + colL]);
    for (int kb = 0; kb < KB; ++kb) {
      bf16x16 a = load_a(bin, wr, strideK, kb, lane);
      bf16x16 b = load_b(w, kb * 16 + nb, lane);
      acc = wmma_bf16(a, b, acc);
    }
#pragma unroll
    for (int r = 0; r < 8; ++r) {
      float v = acc[r];
      v = v > 0.0f ? v : 0.0f;                       // ReLU
      bout[(rbase + r) * DSTR + nb * 16 + colL] = (__bf16)v;
    }
  }
}

__global__ __launch_bounds__(256, 1)
void lstm_sdf_kernel(const float* __restrict__ input,
                     const __bf16* __restrict__ wih,
                     const __bf16* __restrict__ whh,
                     const __bf16* __restrict__ waff,
                     const float* __restrict__ bias_gates,
                     const float* __restrict__ b_aff,
                     const __bf16* __restrict__ w0, const float* __restrict__ b0,
                     const __bf16* __restrict__ w1, const float* __restrict__ b1,
                     const __bf16* __restrict__ w2, const float* __restrict__ b2,
                     const __bf16* __restrict__ w3, const float* __restrict__ b3,
                     const __bf16* __restrict__ w4, const float* __restrict__ b4,
                     const __bf16* __restrict__ w5, const float* __restrict__ b5,
                     float* __restrict__ out) {
  extern __shared__ char smem[];
  __bf16* xcat = (__bf16*)(smem);                                 // [64][XSTR]
  __bf16* hbuf = (__bf16*)(smem + XCAT_B);                        // [64][HSTR]
  float*  cbuf = (float*) (smem + XCAT_B + HBUF_B);               // [64][CSTR]
  float*  xyz  = (float*) (smem + XCAT_B + HBUF_B + CBUF_B);      // [64][3]
  float*  scr  = (float*) (smem + XCAT_B + HBUF_B + CBUF_B + XYZ_B);

  const int tid   = threadIdx.x;
  const int lane  = tid & 31;
  const int wave  = tid >> 5;         // 0..7
  const int rw    = wave >> 1;        // row-slice 0..3
  const int half  = wave & 1;         // column half 0..1
  const int wr    = rw * 16;          // this wave's first row
  const int colL  = lane & 15;
  const int rbase = wr + ((lane & 16) ? 8 : 0);
  const int rowg0 = blockIdx.x * MROWS;

  // ---- init: load xcat (code + xyz, K-padded with zeros), zero h and c ----
  for (int idx = tid; idx < MROWS * XSTR; idx += blockDim.x) {
    int r = idx / XSTR, c = idx % XSTR;
    float v = (c < DDIM) ? input[(size_t)(rowg0 + r) * DDIM + c] : 0.0f;
    xcat[idx] = (__bf16)v;
  }
  for (int idx = tid; idx < MROWS * HSTR; idx += blockDim.x)
    hbuf[idx] = (__bf16)0.0f;
  for (int idx = tid; idx < MROWS * CSTR; idx += blockDim.x)
    cbuf[idx] = 0.0f;
  for (int idx = tid; idx < MROWS * 3; idx += blockDim.x) {
    int r = idx / 3, c = idx % 3;
    xyz[r * 3 + c] = input[(size_t)(rowg0 + r) * DDIM + 256 + c];
  }
  __syncthreads();

  // ------------------------- 8 recurrent steps -----------------------------
#pragma unroll 1
  for (int step = 0; step < NSTEPS; ++step) {
#pragma unroll 1
    for (int nn = 0; nn < 16; ++nn) {             // this wave's column tiles
      const int nt = half * 16 + nn;              // global tile 0..31 (H=512)
      f32x8 acc[4];
#pragma unroll
      for (int q = 0; q < 4; ++q)
        acc[q] = splat8(bias_gates[q * 512 + nt * 16 + colL]);

      // x-part: K = 288 (padded 259); A shared by the 4 gates
#pragma unroll 1
      for (int kb = 0; kb < 9; ++kb) {
        bf16x16 a = load_a(xcat, wr, XSTR, kb, lane);
#pragma unroll
        for (int q = 0; q < 4; ++q) {
          bf16x16 b = load_b(wih, kb * 128 + q * 32 + nt, lane);
          acc[q] = wmma_bf16(a, b, acc[q]);
        }
      }
      // h-part: K = 512
#pragma unroll 1
      for (int kb = 0; kb < 16; ++kb) {
        bf16x16 a = load_a(hbuf, wr, HSTR, kb, lane);
#pragma unroll
        for (int q = 0; q < 4; ++q) {
          bf16x16 b = load_b(whh, kb * 128 + q * 32 + nt, lane);
          acc[q] = wmma_bf16(a, b, acc[q]);
        }
      }

      // LSTM cell update for this 16x16 tile (C-layout -> LDS rows)
      const int col = nt * 16 + colL;
#pragma unroll
      for (int r = 0; r < 8; ++r) {
        const int row = rbase + r;
        float cv = cbuf[row * CSTR + col];
        float iv = sigm(acc[0][r]);
        float fv = sigm(acc[1][r]);
        float gv = tanhf(acc[2][r]);
        float ov = sigm(acc[3][r]);
        cv = fv * cv + iv * gv;
        cbuf[row * CSTR + col] = cv;
        hbuf[row * HSTR + col] = (__bf16)(ov * tanhf(cv));
      }
    }
    __syncthreads();   // h fully written (both halves) before affine head

    // ---- affine head (half-0 waves): a = h @ W_aff.T + b_aff, then xyz ----
    if (half == 0) {
      f32x8 aacc = splat8((colL < 6) ? b_aff[colL] : 0.0f);
#pragma unroll 1
      for (int kb = 0; kb < 16; ++kb) {
        bf16x16 a = load_a(hbuf, wr, HSTR, kb, lane);
        bf16x16 b = load_b(waff, kb, lane);
        aacc = wmma_bf16(a, b, aacc);
      }
      float* sw = scr + rw * 256;
#pragma unroll
      for (int r = 0; r < 8; ++r)
        sw[(((lane & 16) ? 8 : 0) + r) * 16 + colL] = aacc[r];

      // xyz <- a[3:] + (1 + a[:3]) * xyz   (one lane per row, same wave)
      if (lane < 16) {
        const int row = wr + lane;
        float a0 = sw[lane * 16 + 0], a1 = sw[lane * 16 + 1];
        float a2 = sw[lane * 16 + 2], a3 = sw[lane * 16 + 3];
        float a4 = sw[lane * 16 + 4], a5 = sw[lane * 16 + 5];
        float x = xyz[row * 3 + 0], y = xyz[row * 3 + 1], z = xyz[row * 3 + 2];
        x = a3 + (1.0f + a0) * x;
        y = a4 + (1.0f + a1) * y;
        z = a5 + (1.0f + a2) * z;
        xyz[row * 3 + 0] = x; xyz[row * 3 + 1] = y; xyz[row * 3 + 2] = z;
        __bf16* xr = xcat + row * XSTR + 256;
        xr[0] = (__bf16)x; xr[1] = (__bf16)y; xr[2] = (__bf16)z;
      }
    }
    __syncthreads();
  }

  // ------------------------------ decoder ----------------------------------
  // p buffer: 64 x PSTR bf16 (cols 0..2 = xyz, rest zero); reuses xcat space
  __bf16* bufX = xcat;
  for (int idx = tid; idx < MROWS * PSTR; idx += blockDim.x) {
    int r = idx / PSTR, c = idx % PSTR;
    bufX[idx] = (__bf16)((c < 3) ? xyz[r * 3 + c] : 0.0f);
  }
  __syncthreads();

  __bf16* bufA = hbuf;            // 64 x DSTR (fits in h region)
  __bf16* bufB = (__bf16*)cbuf;   // 64 x DSTR (fits in c region)
  const int nb0 = half * 8;       // each half-wave: 8 of 16 output tiles

  decoder_layer(bufX, PSTR, 1, w0, b0, bufA, nb0, lane, wr, rbase, colL);
  __syncthreads();
  decoder_layer(bufA, DSTR, 8, w1, b1, bufB, nb0, lane, wr, rbase, colL);
  __syncthreads();
  decoder_layer(bufB, DSTR, 8, w2, b2, bufA, nb0, lane, wr, rbase, colL);
  __syncthreads();
  decoder_layer(bufA, DSTR, 8, w3, b3, bufB, nb0, lane, wr, rbase, colL);
  __syncthreads();
  decoder_layer(bufB, DSTR, 8, w4, b4, bufA, nb0, lane, wr, rbase, colL);
  __syncthreads();

  // final 256 -> 1 (output column 0 of a padded 16-wide tile), half-0 waves
  if (half == 0) {
    f32x8 acc = splat8(b5[0]);
    for (int kb = 0; kb < 8; ++kb) {
      bf16x16 a = load_a(bufA, wr, DSTR, kb, lane);
      bf16x16 b = load_b(w5, kb, lane);
      acc = wmma_bf16(a, b, acc);
    }
    if (colL == 0) {
#pragma unroll
      for (int r = 0; r < 8; ++r)
        out[rowg0 + rbase + r] = acc[r];
    }
  }
}

// ---------------------------------------------------------------------------
// Host-side launch
// ---------------------------------------------------------------------------
extern "C" void kernel_launch(void* const* d_in, const int* in_sizes, int n_in,
                              void* d_out, int out_size, void* d_ws,
                              size_t ws_size, hipStream_t stream) {
  const float* input = (const float*)d_in[0];
  const float* W_ih  = (const float*)d_in[1];
  const float* b_ih  = (const float*)d_in[2];
  const float* W_hh  = (const float*)d_in[3];
  const float* b_hh  = (const float*)d_in[4];
  const float* W_aff = (const float*)d_in[5];
  const float* b_aff = (const float*)d_in[6];
  const float* v0 = (const float*)d_in[7],  *g0 = (const float*)d_in[8],
             * b0 = (const float*)d_in[9];
  const float* v1 = (const float*)d_in[10], *g1 = (const float*)d_in[11],
             * b1 = (const float*)d_in[12];
  const float* v2 = (const float*)d_in[13], *g2 = (const float*)d_in[14],
             * b2 = (const float*)d_in[15];
  const float* v3 = (const float*)d_in[16], *g3 = (const float*)d_in[17],
             * b3 = (const float*)d_in[18];
  const float* v4 = (const float*)d_in[19], *g4 = (const float*)d_in[20],
             * b4 = (const float*)d_in[21];
  const float* W5 = (const float*)d_in[22], *b5 = (const float*)d_in[23];
  float* out = (float*)d_out;
  char* ws = (char*)d_ws;

  // workspace carve-out (256 B aligned)
  size_t off = 0;
  auto carve = [&](size_t bytes) {
    size_t o = off;
    off = (off + bytes + 255) & ~(size_t)255;
    return o;
  };
  const size_t o_wih  = carve((size_t)9  * 128 * 512 * 2);
  const size_t o_whh  = carve((size_t)16 * 128 * 512 * 2);
  const size_t o_waff = carve((size_t)16 * 1   * 512 * 2);
  const size_t o_w0   = carve((size_t)1  * 16  * 512 * 2);
  const size_t o_w1   = carve((size_t)8  * 16  * 512 * 2);
  const size_t o_w2   = carve((size_t)8  * 16  * 512 * 2);
  const size_t o_w3   = carve((size_t)8  * 16  * 512 * 2);
  const size_t o_w4   = carve((size_t)8  * 16  * 512 * 2);
  const size_t o_w5   = carve((size_t)8  * 1   * 512 * 2);
  const size_t o_bg   = carve(2048 * 4);
  const size_t o_s    = carve(5 * 256 * 4);
  (void)ws_size; (void)n_in; (void)in_sizes; (void)out_size;

  float* scales = (float*)(ws + o_s);

  // --- precompute: weight-norm scales, bf16 fragments, fused gate bias ---
  rownorm_kernel<<<5 * 256, 32, 0, stream>>>(v0, g0, v1, g1, v2, g2,
                                             v3, g3, v4, g4, scales);

  auto frag = [&](const float* W, const float* sc, size_t o,
                  int N, int K, int KB, int NB) {
    int total = KB * NB * 512;
    fragw_kernel<<<(total + 255) / 256, 256, 0, stream>>>(
        W, sc, (__bf16*)(ws + o), N, K, KB, NB);
  };
  frag(W_ih,  nullptr,      o_wih,  2048, 259, 9,  128);
  frag(W_hh,  nullptr,      o_whh,  2048, 512, 16, 128);
  frag(W_aff, nullptr,      o_waff, 6,    512, 16, 1);
  frag(v0,    scales + 0,   o_w0,   256,  3,   1,  16);
  frag(v1,    scales + 256, o_w1,   256,  256, 8,  16);
  frag(v2,    scales + 512, o_w2,   256,  256, 8,  16);
  frag(v3,    scales + 768, o_w3,   256,  256, 8,  16);
  frag(v4,    scales + 1024,o_w4,   256,  256, 8,  16);
  frag(W5,    nullptr,      o_w5,   1,    256, 8,  1);

  biasgate_kernel<<<8, 256, 0, stream>>>(b_ih, b_hh, (float*)(ws + o_bg));

  // --- fused LSTM + decoder: 1024 workgroups x 8 waves (2 per SIMD32) ---
  lstm_sdf_kernel<<<65536 / MROWS, 256, SMEM_TOTAL, stream>>>(
      input,
      (const __bf16*)(ws + o_wih), (const __bf16*)(ws + o_whh),
      (const __bf16*)(ws + o_waff),
      (const float*)(ws + o_bg), b_aff,
      (const __bf16*)(ws + o_w0), b0,
      (const __bf16*)(ws + o_w1), b1,
      (const __bf16*)(ws + o_w2), b2,
      (const __bf16*)(ws + o_w3), b3,
      (const __bf16*)(ws + o_w4), b4,
      (const __bf16*)(ws + o_w5), b5,
      out);
}
